// TransformerBlock_39015482916995
// MI455X (gfx1250) — matrix operand order repaired and checked by reading, not compiled, not run on hardware
//
#include <hip/hip_runtime.h>

// ---------------------------------------------------------------------------
// Transformer block forward for gfx1250 (CDNA5, wave32, WMMA).
// GEMMs: v_wmma_f32_16x16x32_bf16, 32x64 wave tiles (8 WMMA per K-step).
// Attention: flash-style online softmax; K tile staged via CDNA5 async
// global->LDS copies (ASYNCcnt), V tile transpose-staged with 128b loads.
// ---------------------------------------------------------------------------

typedef __attribute__((ext_vector_type(16))) __bf16 bf16x16;
typedef __attribute__((ext_vector_type(8)))  __bf16 bf16x8;
typedef __attribute__((ext_vector_type(8)))  float  f32x8;

#define BQ 2
#define TQ 2048
#define DQ 1024
#define HQ 16
#define DHQ 64
#define DFFQ 4096

// ---- CDNA5 async global->LDS copy (tracked by ASYNCcnt) --------------------
__device__ inline void async_g2l_b128(unsigned lds_byte_addr, const void* gaddr) {
    asm volatile("global_load_async_to_lds_b128 %0, %1, off"
                 :: "v"(lds_byte_addr), "v"(gaddr)
                 : "memory");
}
__device__ inline void wait_async0() {
    asm volatile("s_wait_asynccnt 0" ::: "memory");
}

// ---- WMMA fragment helpers (contiguous k order, wave32) ----------
// A matrix 16x32 bf16 (M x K), row-major with leading dim lda:
//   lanes 0-15 : row = lane,     K = 0..15
//   lanes 16-31: row = lane-16,  K = 16..31
template <typename PT>
__device__ inline bf16x16 load_a_frag(PT A, int lda, int row0, int k0, int lane) {
    int row = row0 + (lane & 15);
    int kb  = (lane & 16) ? 16 : 0;
    return *(const bf16x16*)(A + (size_t)row * lda + k0 + kb);
}

// B matrix 32x16 bf16 (K x N) supplied TRANSPOSED (Bt is N x K row-major):
//   lanes 0-15 : col = lane,     K = 0..15
//   lanes 16-31: col = lane-16,  K = 16..31
template <typename PT>
__device__ inline bf16x16 load_b_frag(PT Bt, int ldb, int n0, int k0, int lane) {
    int col = n0 + (lane & 15);
    int kb  = (lane & 16) ? 16 : 0;
    return *(const bf16x16*)(Bt + (size_t)col * ldb + k0 + kb);
}

__device__ inline f32x8 wmma_bf16(bf16x16 a, bf16x16 b, f32x8 c) {
    // (neg_a, A, neg_b, B, c_mod, C, reuse_a, reuse_b)
    return __builtin_amdgcn_wmma_f32_16x16x32_bf16(false, a, false, b,
                                                   (short)0, c, false, false);
}

// ---- fp32 -> bf16 transposed weight conversion -----------------------------
__global__ void cvt_t_kernel(const float* __restrict__ w, __bf16* __restrict__ wt,
                             int K, int N) {
    size_t idx = (size_t)blockIdx.x * blockDim.x + threadIdx.x;
    size_t total = (size_t)K * N;
    if (idx >= total) return;
    int k = (int)(idx / N);
    int n = (int)(idx % N);
    wt[(size_t)n * K + k] = (__bf16)w[idx];
}

// ---- LayerNorm (one row per block), bf16 output ----------------------------
__global__ __launch_bounds__(256)
void ln_kernel(const float* __restrict__ x, const float* __restrict__ g,
               const float* __restrict__ b, __bf16* __restrict__ out, int D) {
    int row = blockIdx.x;
    const float* xr = x + (size_t)row * D;
    float s = 0.f, ss = 0.f;
    for (int i = threadIdx.x; i < D; i += 256) {
        float v = xr[i];
        s += v; ss += v * v;
    }
    __shared__ float sh[512];
    sh[threadIdx.x] = s;
    sh[threadIdx.x + 256] = ss;
    __syncthreads();
    for (int o = 128; o > 0; o >>= 1) {
        if (threadIdx.x < (unsigned)o) {
            sh[threadIdx.x]       += sh[threadIdx.x + o];
            sh[threadIdx.x + 256] += sh[threadIdx.x + 256 + o];
        }
        __syncthreads();
    }
    float mean = sh[0] / D;
    float var  = sh[256] / D - mean * mean;
    float rstd = rsqrtf(var + 1e-5f);
    for (int i = threadIdx.x; i < D; i += 256) {
        float v = (xr[i] - mean) * rstd * g[i] + b[i];
        out[(size_t)row * D + i] = (__bf16)v;
    }
}

// ---- Generic bf16 GEMM: C = act(scale * A @ Bt^T + bias) + residual --------
// A: (M x K) bf16 row-major.  Bt: (N x K) bf16 row-major (i.e. B transposed).
// Block tile 128x128, 8 waves, wave tile 32x64 (2x4 WMMA 16x16 tiles).
__global__ __launch_bounds__(256)
void gemm_kernel(const __bf16* __restrict__ A, const __bf16* __restrict__ Bt,
                 const float* __restrict__ bias, const float* __restrict__ residual,
                 float* __restrict__ outF, __bf16* __restrict__ outB,
                 int M, int N, int K, float scale, int act_gelu) {
    const int lane = threadIdx.x & 31;
    const int w    = threadIdx.x >> 5;
    const int wm   = w & 3;   // 0..3 along M
    const int wn   = w >> 2;  // 0..1 along N
    const int row0 = blockIdx.y * 128 + wm * 32;
    const int col0 = blockIdx.x * 128 + wn * 64;

    f32x8 acc[2][4] = {};

    for (int k0 = 0; k0 < K; k0 += 32) {
        if ((k0 & 255) == 0) {  // gfx1250 global_prefetch path
            __builtin_prefetch(A  + (size_t)row0 * K + k0 + 256, 0, 0);
            __builtin_prefetch(Bt + (size_t)col0 * K + k0 + 256, 0, 0);
        }
        bf16x16 a0 = load_a_frag(A, K, row0,      k0, lane);
        bf16x16 a1 = load_a_frag(A, K, row0 + 16, k0, lane);
#pragma unroll
        for (int j = 0; j < 4; j++) {
            bf16x16 bj = load_b_frag(Bt, K, col0 + j * 16, k0, lane);
            acc[0][j] = wmma_bf16(a0, bj, acc[0][j]);
            acc[1][j] = wmma_bf16(a1, bj, acc[1][j]);
        }
    }

    // C/D layout: lanes 0-15 col=lane rows r; lanes 16-31 col=lane-16 rows r+8
    const int rbase = (lane & 16) ? 8 : 0;
    const int cl    = lane & 15;
#pragma unroll
    for (int i = 0; i < 2; i++) {
#pragma unroll
        for (int j = 0; j < 4; j++) {
            int col  = col0 + j * 16 + cl;
            float bs = bias ? bias[col] : 0.f;
#pragma unroll
            for (int r = 0; r < 8; r++) {
                int row = row0 + i * 16 + rbase + r;
                float v = acc[i][j][r] * scale + bs;
                if (act_gelu) v = 0.5f * v * (1.f + erff(v * 0.70710678118654752f));
                size_t o = (size_t)row * N + col;
                if (residual) v += residual[o];
                if (outF) outF[o] = v;
                if (outB) outB[o] = (__bf16)v;
            }
        }
    }
}

// ---- Flash attention: per (b, h, 64-query tile); 4 waves x 16 rows ---------
__global__ __launch_bounds__(128)
void attn_kernel(const __bf16* __restrict__ Qg, const __bf16* __restrict__ Kg,
                 const __bf16* __restrict__ Vg, __bf16* __restrict__ Og) {
    const int D  = DQ;
    const int lane = threadIdx.x & 31;
    const int w    = threadIdx.x >> 5;
    const int qt = blockIdx.x;
    const int h  = blockIdx.y;
    const int b  = blockIdx.z;
    const size_t base = (size_t)b * TQ * D + (size_t)h * DHQ;  // + t*D + d
    const int q0 = qt * 64 + w * 16;

    __shared__ __bf16 Ks[64][64];      // K tile as-is (key-major) => Bt for Q@K^T
    __shared__ __bf16 Vt[64][64];      // V tile transposed (dh-major) => Bt for P@V
    __shared__ __bf16 Ps[4][16][64];   // per-wave probability tile (A layout source)

    bf16x16 aq0 = load_a_frag(Qg + base, D, q0, 0,  lane);
    bf16x16 aq1 = load_a_frag(Qg + base, D, q0, 32, lane);

    f32x8 o_acc[4] = {};
    float m_r[8], l_r[8];
#pragma unroll
    for (int r = 0; r < 8; r++) { m_r[r] = -3.0e38f; l_r[r] = 0.f; }

    const int rbase = (lane & 16) ? 8 : 0;
    const int cl    = lane & 15;

    for (int kt = 0; kt <= qt; kt++) {
        const int k0g = kt * 64;
        __syncthreads();

        // --- stage K tile via async global->LDS b128 (8KB, 512 lane-ops) ---
#pragma unroll
        for (int i = 0; i < 4; i++) {
            int idx = threadIdx.x + i * 128;       // 0..511
            int kk  = idx >> 3;                    // key row 0..63
            int seg = idx & 7;                     // 16B segment 0..7
            const __bf16* g = Kg + base + (size_t)(k0g + kk) * D + seg * 8;
            unsigned l = (unsigned)(uintptr_t)&Ks[kk][seg * 8]; // flat[31:0] = LDS addr
            async_g2l_b128(l, g);
        }
        // --- stage V tile transposed (vectorized 16B global loads) ---------
#pragma unroll
        for (int i = 0; i < 4; i++) {
            int idx = threadIdx.x + i * 128;
            int kk  = idx >> 3;
            int seg = idx & 7;
            bf16x8 vv = *(const bf16x8*)(Vg + base + (size_t)(k0g + kk) * D + seg * 8);
#pragma unroll
            for (int e = 0; e < 8; e++) Vt[seg * 8 + e][kk] = vv[e];
        }
        wait_async0();
        __syncthreads();

        // scores: 16 x 64, four 16x16 tiles, K-dim = DH = 64
        f32x8 s[4] = {};
#pragma unroll
        for (int j = 0; j < 4; j++) {
            bf16x16 b0 = load_b_frag(&Ks[0][0], 64, j * 16, 0,  lane);
            bf16x16 b1 = load_b_frag(&Ks[0][0], 64, j * 16, 32, lane);
            s[j] = wmma_bf16(aq0, b0, s[j]);
            s[j] = wmma_bf16(aq1, b1, s[j]);
        }

        // online softmax per row (rows live across 16-lane half-waves)
#pragma unroll
        for (int r = 0; r < 8; r++) {
            int row = q0 + rbase + r;
            float sv[4];
#pragma unroll
            for (int j = 0; j < 4; j++) {
                int key = k0g + j * 16 + cl;
                float xv = s[j][r] * 0.125f;          // 1/sqrt(64)
                if (key > row) xv = -3.0e38f;          // causal mask
                sv[j] = xv;
            }
            float mx = fmaxf(fmaxf(sv[0], sv[1]), fmaxf(sv[2], sv[3]));
#pragma unroll
            for (int off = 8; off >= 1; off >>= 1)
                mx = fmaxf(mx, __shfl_xor(mx, off, 32));
            float mnew  = fmaxf(m_r[r], mx);
            float alpha = __expf(m_r[r] - mnew);
            float psum = 0.f;
#pragma unroll
            for (int j = 0; j < 4; j++) {
                float pv = __expf(sv[j] - mnew);
                psum += pv;
                Ps[w][rbase + r][j * 16 + cl] = (__bf16)pv;
            }
#pragma unroll
            for (int off = 8; off >= 1; off >>= 1)
                psum += __shfl_xor(psum, off, 32);
            l_r[r] = l_r[r] * alpha + psum;
            m_r[r] = mnew;
#pragma unroll
            for (int t = 0; t < 4; t++) o_acc[t][r] *= alpha;
        }

        // O += P @ V  (A = P 16x64 from LDS, Bt = Vt)
        bf16x16 ap0 = load_a_frag(&Ps[w][0][0], 64, 0, 0,  lane);
        bf16x16 ap1 = load_a_frag(&Ps[w][0][0], 64, 0, 32, lane);
#pragma unroll
        for (int t = 0; t < 4; t++) {
            bf16x16 bv0 = load_b_frag(&Vt[0][0], 64, t * 16, 0,  lane);
            bf16x16 bv1 = load_b_frag(&Vt[0][0], 64, t * 16, 32, lane);
            o_acc[t] = wmma_bf16(ap0, bv0, o_acc[t]);
            o_acc[t] = wmma_bf16(ap1, bv1, o_acc[t]);
        }
    }

#pragma unroll
    for (int t = 0; t < 4; t++) {
#pragma unroll
        for (int r = 0; r < 8; r++) {
            int row = q0 + rbase + r;
            float v = o_acc[t][r] / l_r[r];
            Og[base + (size_t)row * D + t * 16 + cl] = (__bf16)v;
        }
    }
}

// ---------------------------------------------------------------------------
extern "C" void kernel_launch(void* const* d_in, const int* in_sizes, int n_in,
                              void* d_out, int out_size, void* d_ws, size_t ws_size,
                              hipStream_t stream) {
    (void)in_sizes; (void)n_in; (void)out_size; (void)ws_size;
    const int B = BQ, T = TQ, D = DQ, H = HQ, DFF = DFFQ;
    const int M = B * T;  // 4096 token rows

    const float* x      = (const float*)d_in[0];
    /* d_in[1] = causal_mask (bool) — causality computed analytically */
    const float* gamma1 = (const float*)d_in[2];
    const float* beta1  = (const float*)d_in[3];
    const float* wq     = (const float*)d_in[4];
    const float* wk     = (const float*)d_in[5];
    const float* wv     = (const float*)d_in[6];
    const float* wo     = (const float*)d_in[7];
    const float* gamma2 = (const float*)d_in[8];
    const float* beta2  = (const float*)d_in[9];
    const float* w1     = (const float*)d_in[10];
    const float* b1     = (const float*)d_in[11];
    const float* w2     = (const float*)d_in[12];
    const float* b2     = (const float*)d_in[13];
    float* out = (float*)d_out;

    // workspace carve-out (256B aligned)
    char* p = (char*)d_ws;
    auto take = [&](size_t bytes) -> char* {
        char* r = p;
        p += (bytes + 255) & ~(size_t)255;
        return r;
    };
    __bf16* wqT  = (__bf16*)take((size_t)D * D   * 2);
    __bf16* wkT  = (__bf16*)take((size_t)D * D   * 2);
    __bf16* wvT  = (__bf16*)take((size_t)D * D   * 2);
    __bf16* woT  = (__bf16*)take((size_t)D * D   * 2);
    __bf16* w1T  = (__bf16*)take((size_t)D * DFF * 2);
    __bf16* w2T  = (__bf16*)take((size_t)DFF * D * 2);
    __bf16* h_bf = (__bf16*)take((size_t)M * D   * 2);
    __bf16* q_bf = (__bf16*)take((size_t)M * D   * 2);
    __bf16* k_bf = (__bf16*)take((size_t)M * D   * 2);
    __bf16* v_bf = (__bf16*)take((size_t)M * D   * 2);
    __bf16* a_bf = (__bf16*)take((size_t)M * D   * 2);
    float*  x1   = (float*)take((size_t)M * D    * 4);
    __bf16* f1   = (__bf16*)take((size_t)M * DFF * 2);

    // 1) weight convert + transpose (f32 -> bf16, N x K)
    cvt_t_kernel<<<(D * D + 255) / 256, 256, 0, stream>>>(wq, wqT, D, D);
    cvt_t_kernel<<<(D * D + 255) / 256, 256, 0, stream>>>(wk, wkT, D, D);
    cvt_t_kernel<<<(D * D + 255) / 256, 256, 0, stream>>>(wv, wvT, D, D);
    cvt_t_kernel<<<(D * D + 255) / 256, 256, 0, stream>>>(wo, woT, D, D);
    cvt_t_kernel<<<(D * DFF + 255) / 256, 256, 0, stream>>>(w1, w1T, D, DFF);
    cvt_t_kernel<<<(DFF * D + 255) / 256, 256, 0, stream>>>(w2, w2T, DFF, D);

    // 2) pre-LN 1
    ln_kernel<<<M, 256, 0, stream>>>(x, gamma1, beta1, h_bf, D);

    // 3) Q/K/V projections (Q pre-scaled by 1/sqrt(Dh) = 0.125)
    dim3 gProj(D / 128, M / 128);
    gemm_kernel<<<gProj, 256, 0, stream>>>(h_bf, wqT, nullptr, nullptr,
                                           nullptr, q_bf, M, D, D, 0.125f, 0);
    gemm_kernel<<<gProj, 256, 0, stream>>>(h_bf, wkT, nullptr, nullptr,
                                           nullptr, k_bf, M, D, D, 1.0f, 0);
    gemm_kernel<<<gProj, 256, 0, stream>>>(h_bf, wvT, nullptr, nullptr,
                                           nullptr, v_bf, M, D, D, 1.0f, 0);

    // 4) flash attention (causal)
    dim3 gAttn(T / 64, H, B);
    attn_kernel<<<gAttn, 128, 0, stream>>>(q_bf, k_bf, v_bf, a_bf);

    // 5) output projection + residual: x1 = x + attn @ Wo
    gemm_kernel<<<gProj, 256, 0, stream>>>(a_bf, woT, nullptr, x,
                                           x1, nullptr, M, D, D, 1.0f, 0);

    // 6) pre-LN 2
    ln_kernel<<<M, 256, 0, stream>>>(x1, gamma2, beta2, h_bf, D);

    // 7) FFN up + GELU (exact)
    dim3 gFfn1(DFF / 128, M / 128);
    gemm_kernel<<<gFfn1, 256, 0, stream>>>(h_bf, w1T, b1, nullptr,
                                           nullptr, f1, M, DFF, D, 1.0f, 1);

    // 8) FFN down + bias + residual -> final output
    gemm_kernel<<<gProj, 256, 0, stream>>>(f1, w2T, b2, x1,
                                           out, nullptr, M, D, DFF, 1.0f, 0);
}
